// Attention_38405597561417
// MI455X (gfx1250) — compile-verified, matrix operand order checked by reading, not run on hardware
//
#include <hip/hip_runtime.h>
#include <hip/hip_bf16.h>
#include <math.h>

// ---------------------------------------------------------------------------
// MHA block for MI455X (gfx1250, wave32, WMMA).
// Precision: bf16 operands + f32 accumulation via v_wmma_f32_16x16x32_bf16.
// Data movement: GLOBAL_LOAD_ASYNC_TO_LDS_B128 (ASYNCcnt) for bf16 staging,
// global_prefetch for the next K/V block (192MB L2).
// Workspace layout (bytes): Qb[0,8M) Kb[8M,16M) Vb[16M,24M) Ob[24M,32M)
// ---------------------------------------------------------------------------

#define S_LEN 4096
#define DMODEL 1024
#define NHEADS 16
#define DKH 64

typedef __attribute__((ext_vector_type(16))) __bf16 v16bf;
typedef __attribute__((ext_vector_type(8)))  float  v8f;

static __device__ __forceinline__ unsigned short f32_to_bf16u(float f) {
    unsigned int u = __float_as_uint(f);
    unsigned int r = u + 0x7FFFu + ((u >> 16) & 1u);   // round-to-nearest-even
    return (unsigned short)(r >> 16);
}

static __device__ __forceinline__ __bf16 us_bf(unsigned short u) {
    union { unsigned short s; __bf16 b; } c; c.s = u; return c.b;
}

// Shared-aperture flat addresses carry the LDS byte address in addr[31:0]
// (ISA 10.2 aperture mapping), so truncation yields the DS address.
static __device__ __forceinline__ unsigned lds_addr(const void* p) {
    return (unsigned)(unsigned long long)p;
}

// GLOBAL_LOAD_ASYNC_TO_LDS_B128, GV mode: LDS[vdst] = MEM[vaddr64]. ASYNCcnt.
static __device__ __forceinline__ void async_ld_b128(unsigned lds_byte,
                                                     const void* gptr) {
    asm volatile("global_load_async_to_lds_b128 %0, %1, off"
                 :: "v"(lds_byte), "v"((unsigned long long)gptr)
                 : "memory");
}

static __device__ __forceinline__ void wait_async0() {
#if __has_builtin(__builtin_amdgcn_s_wait_asynccnt)
    __builtin_amdgcn_s_wait_asynccnt(0);
#else
    asm volatile("s_wait_asynccnt 0x0" ::: "memory");
#endif
}

// A/B fragment from a row-major [row][k] LDS tile (ISA 7.12.2, 16-bit 16x32):
// lane-half h=lane>>4 owns K in {8h..8h+7, 16+8h..16+8h+7}; row = lane&15.
static __device__ __forceinline__ v16bf frag_rowmajor(const unsigned short* p,
                                                      int row_base, int k_base, int ld) {
    const int lane = threadIdx.x & 31;
    const int h = lane >> 4;
    const int r = lane & 15;
    const unsigned short* row = p + (row_base + r) * ld + k_base;
    v16bf out;
#pragma unroll
    for (int v = 0; v < 8; ++v) {
        const int kk = 16 * (v >> 2) + 8 * h + 2 * (v & 3);
        out[2 * v]     = us_bf(row[kk]);
        out[2 * v + 1] = us_bf(row[kk + 1]);
    }
    return out;
}

// B fragment when source is stored k-major [k][n] (used for P@V with V[j][dk]).
static __device__ __forceinline__ v16bf frag_kmajor(const unsigned short* p,
                                                    int n_base, int k_base, int ld) {
    const int lane = threadIdx.x & 31;
    const int h = lane >> 4;
    const int n = lane & 15;
    v16bf out;
#pragma unroll
    for (int v = 0; v < 8; ++v) {
        const int kk = 16 * (v >> 2) + 8 * h + 2 * (v & 3);
        out[2 * v]     = us_bf(p[(k_base + kk)     * ld + n_base + n]);
        out[2 * v + 1] = us_bf(p[(k_base + kk + 1) * ld + n_base + n]);
    }
    return out;
}

static __device__ __forceinline__ v8f wmma_bf16(v16bf a, v16bf b, v8f c) {
    return __builtin_amdgcn_wmma_f32_16x16x32_bf16(false, a, false, b,
                                                   (short)0, c, false, false);
}

// ---------------------------------------------------------------------------
// Projection GEMM: Y[4096,1024] = X[4096,1024] @ W[1024,1024]^T + bias.
// Block: 128(M) x 128(N) tile, 256 threads = 8 waves (2x4), each wave 64x32.
// X is f32 (Xf) or bf16-in-ws (Xb). Outputs: Yb (bf16 [s,d]),
// Yhead (f32 [head][s][dk]) for key_/value, Yplain (f32 [s,d]) for final out.
// ---------------------------------------------------------------------------
__global__ __launch_bounds__(256)
void proj_kernel(const float* __restrict__ Xf, const unsigned short* __restrict__ Xb,
                 const float* __restrict__ W, const float* __restrict__ bias,
                 unsigned short* __restrict__ Yb, float* __restrict__ Yhead,
                 float* __restrict__ Yplain) {
    __shared__ unsigned short As[128 * 32];
    __shared__ unsigned short Ws[128 * 32];

    const int mbase = blockIdx.x * 128;
    const int nbase = blockIdx.y * 128;
    const int tid   = threadIdx.x;
    const int wid   = tid >> 5;
    const int lane  = tid & 31;
    const int h     = lane >> 4;
    const int wm    = (wid >> 2) * 64;   // 0 or 64
    const int wn    = (wid & 3) * 32;    // 0,32,64,96

    v8f acc[4][2];
#pragma unroll
    for (int mi = 0; mi < 4; ++mi)
#pragma unroll
        for (int ni = 0; ni < 2; ++ni) acc[mi][ni] = (v8f)(0.0f);

    const int srow = tid >> 1;            // 0..127
    const int scol = (tid & 1) * 16;      // 0 or 16

    for (int k0 = 0; k0 < DMODEL; k0 += 32) {
        // Stage A (X) 128x32 as bf16.
        if (Xb) {
            // Already bf16: async copy straight into LDS (no VGPR round-trip).
            const unsigned short* g = Xb + (size_t)(mbase + srow) * DMODEL + k0 + scol;
            const unsigned l = lds_addr(&As[srow * 32 + scol]);
            async_ld_b128(l,      g);
            async_ld_b128(l + 16, g + 8);
            wait_async0();
        } else {
            const float4* src = (const float4*)(Xf + (size_t)(mbase + srow) * DMODEL + k0 + scol);
#pragma unroll
            for (int q = 0; q < 4; ++q) {
                float4 f = src[q];
                As[srow * 32 + scol + 4 * q + 0] = f32_to_bf16u(f.x);
                As[srow * 32 + scol + 4 * q + 1] = f32_to_bf16u(f.y);
                As[srow * 32 + scol + 4 * q + 2] = f32_to_bf16u(f.z);
                As[srow * 32 + scol + 4 * q + 3] = f32_to_bf16u(f.w);
            }
        }
        // Stage W 128x32 as bf16 ([n][k] row-major == B^T, exactly what WMMA-B wants).
        {
            const float4* src = (const float4*)(W + (size_t)(nbase + srow) * DMODEL + k0 + scol);
#pragma unroll
            for (int q = 0; q < 4; ++q) {
                float4 f = src[q];
                Ws[srow * 32 + scol + 4 * q + 0] = f32_to_bf16u(f.x);
                Ws[srow * 32 + scol + 4 * q + 1] = f32_to_bf16u(f.y);
                Ws[srow * 32 + scol + 4 * q + 2] = f32_to_bf16u(f.z);
                Ws[srow * 32 + scol + 4 * q + 3] = f32_to_bf16u(f.w);
            }
        }
        __syncthreads();

        v16bf bfr[2];
#pragma unroll
        for (int ni = 0; ni < 2; ++ni)
            bfr[ni] = frag_rowmajor(Ws, wn + ni * 16, 0, 32);
#pragma unroll
        for (int mi = 0; mi < 4; ++mi) {
            v16bf afr = frag_rowmajor(As, wm + mi * 16, 0, 32);
#pragma unroll
            for (int ni = 0; ni < 2; ++ni)
                acc[mi][ni] = wmma_bf16(afr, bfr[ni], acc[mi][ni]);
        }
        __syncthreads();
    }

    // Epilogue: bias + stores.
#pragma unroll
    for (int ni = 0; ni < 2; ++ni) {
        const int d = nbase + wn + ni * 16 + (lane & 15);
        const float bv = bias[d];
#pragma unroll
        for (int mi = 0; mi < 4; ++mi) {
#pragma unroll
            for (int r = 0; r < 8; ++r) {
                const int mg = mbase + wm + mi * 16 + r + 8 * h;
                const float val = acc[mi][ni][r] + bv;
                if (Yb)     Yb[mg * DMODEL + d] = f32_to_bf16u(val);
                if (Yhead)  Yhead[(d >> 6) * (S_LEN * DKH) + mg * DKH + (d & 63)] = val;
                if (Yplain) Yplain[mg * DMODEL + d] = val;
            }
        }
    }
}

// ---------------------------------------------------------------------------
// Flash attention (causal), one (head, 64-row q-tile) per block.
// 128 threads = 4 waves; wave w owns q rows [w*16, w*16+16).
// K/V blocks staged via GLOBAL_LOAD_ASYNC_TO_LDS_B128.
// ---------------------------------------------------------------------------
__global__ __launch_bounds__(128)
void attn_kernel(const unsigned short* __restrict__ Qb,
                 const unsigned short* __restrict__ Kb,
                 const unsigned short* __restrict__ Vb,
                 unsigned short* __restrict__ Ob) {
    __shared__ unsigned short Qs[64 * 64];
    __shared__ unsigned short Ks[32 * 64];
    __shared__ unsigned short Vs[32 * 64];
    __shared__ unsigned short Ps[4][16 * 32];

    const int qbase = blockIdx.x * 64;
    const int head  = blockIdx.y;
    const int dh0   = head * DKH;
    const int tid   = threadIdx.x;
    const int wid   = tid >> 5;
    const int lane  = tid & 31;
    const int h     = lane >> 4;
    const int m0    = wid * 16;

    // Stage Q tile (64x64 bf16) asynchronously: 32 elements/thread.
    {
        const int row = tid >> 1, col = (tid & 1) * 32;
        const unsigned short* g = Qb + (size_t)(qbase + row) * DMODEL + dh0 + col;
        const unsigned l = lds_addr(&Qs[row * 64 + col]);
#pragma unroll
        for (int q = 0; q < 4; ++q)
            async_ld_b128(l + 16 * q, g + 8 * q);
        wait_async0();
    }
    __syncthreads();

    v16bf aq0 = frag_rowmajor(Qs, m0, 0, 64);
    v16bf aq1 = frag_rowmajor(Qs, m0, 32, 64);

    v8f o[4];
#pragma unroll
    for (int ni = 0; ni < 4; ++ni) o[ni] = (v8f)(0.0f);
    float mrow[8], lrow[8];
#pragma unroll
    for (int r = 0; r < 8; ++r) { mrow[r] = -3.0e38f; lrow[r] = 0.0f; }

    for (int j0 = 0; j0 < qbase + 64; j0 += 32) {
        // Stage K,V blocks (32x64 bf16 each) via async-to-LDS; prefetch next.
        {
            const int row = tid >> 2, col = (tid & 3) * 16;
            const unsigned short* gk = Kb + (size_t)(j0 + row) * DMODEL + dh0 + col;
            const unsigned short* gv = Vb + (size_t)(j0 + row) * DMODEL + dh0 + col;
            const unsigned lk = lds_addr(&Ks[row * 64 + col]);
            const unsigned lv = lds_addr(&Vs[row * 64 + col]);
            async_ld_b128(lk,      gk);
            async_ld_b128(lk + 16, gk + 8);
            async_ld_b128(lv,      gv);
            async_ld_b128(lv + 16, gv + 8);
            if (j0 + 32 < qbase + 64) {          // warm L2/WGP$ for next block
                __builtin_prefetch(gk + 32 * DMODEL, 0, 1);
                __builtin_prefetch(gv + 32 * DMODEL, 0, 1);
            }
            wait_async0();
        }
        __syncthreads();

        // S tile 16x32 = Q(16x64) @ K^T(64x32), two n-groups of 16.
        v8f s[2];
#pragma unroll
        for (int jn = 0; jn < 2; ++jn) {
            s[jn] = (v8f)(0.0f);
            s[jn] = wmma_bf16(aq0, frag_rowmajor(Ks, jn * 16, 0, 64), s[jn]);
            s[jn] = wmma_bf16(aq1, frag_rowmajor(Ks, jn * 16, 32, 64), s[jn]);
        }

        // Scale + causal mask (-3e38 instead of -inf to keep exp finite).
#pragma unroll
        for (int jn = 0; jn < 2; ++jn) {
            const int ng = j0 + jn * 16 + (lane & 15);
#pragma unroll
            for (int r = 0; r < 8; ++r) {
                const int mg = qbase + m0 + r + 8 * h;
                float v = s[jn][r] * 0.125f;
                s[jn][r] = (ng > mg) ? -3.0e38f : v;
            }
        }

        // Online softmax: row reductions stay inside the 16-lane half.
#pragma unroll
        for (int r = 0; r < 8; ++r) {
            float mx = fmaxf(s[0][r], s[1][r]);
#pragma unroll
            for (int d = 1; d < 16; d <<= 1)
                mx = fmaxf(mx, __shfl_xor(mx, d, 32));
            const float mnew = fmaxf(mrow[r], mx);
            const float alpha = expf(mrow[r] - mnew);
            const float p0 = expf(s[0][r] - mnew);
            const float p1 = expf(s[1][r] - mnew);
            s[0][r] = p0; s[1][r] = p1;
            float rs = p0 + p1;
#pragma unroll
            for (int d = 1; d < 16; d <<= 1)
                rs += __shfl_xor(rs, d, 32);
            lrow[r] = lrow[r] * alpha + rs;
            mrow[r] = mnew;
#pragma unroll
            for (int ni = 0; ni < 4; ++ni) o[ni][r] *= alpha;
        }

        // P (f32, C-layout) -> LDS -> A-fragment layout (bf16).
#pragma unroll
        for (int jn = 0; jn < 2; ++jn)
#pragma unroll
            for (int r = 0; r < 8; ++r)
                Ps[wid][(r + 8 * h) * 32 + jn * 16 + (lane & 15)] = f32_to_bf16u(s[jn][r]);
        // Same-wave DS ops are in-order; no workgroup barrier needed here.
        v16bf ap = frag_rowmajor(Ps[wid], 0, 0, 32);

        // O += P(16x32) @ V(32x64); V stored [j][dk] => k-major B fragments.
#pragma unroll
        for (int ni = 0; ni < 4; ++ni)
            o[ni] = wmma_bf16(ap, frag_kmajor(Vs, ni * 16, 0, 64), o[ni]);

        __syncthreads();   // before restaging K/V
    }

    // Normalize and store O as bf16 [s, d] for the output projection.
#pragma unroll
    for (int ni = 0; ni < 4; ++ni) {
        const int d = dh0 + ni * 16 + (lane & 15);
#pragma unroll
        for (int r = 0; r < 8; ++r) {
            const int mg = qbase + m0 + r + 8 * h;
            Ob[mg * DMODEL + d] = f32_to_bf16u(o[ni][r] / lrow[r]);
        }
    }
}

// ---------------------------------------------------------------------------
extern "C" void kernel_launch(void* const* d_in, const int* in_sizes, int n_in,
                              void* d_out, int out_size, void* d_ws, size_t ws_size,
                              hipStream_t stream) {
    (void)in_sizes; (void)n_in; (void)out_size; (void)ws_size;
    const float* q  = (const float*)d_in[0];
    const float* k  = (const float*)d_in[1];
    const float* v  = (const float*)d_in[2];
    const float* Wq = (const float*)d_in[3];
    const float* bq = (const float*)d_in[4];
    const float* Wk = (const float*)d_in[5];
    const float* bk = (const float*)d_in[6];
    const float* Wv = (const float*)d_in[7];
    const float* bv = (const float*)d_in[8];
    const float* Wo = (const float*)d_in[9];
    const float* bo = (const float*)d_in[10];

    float* out      = (float*)d_out;                    // [0, 4096*1024)
    float* key_out  = out + (size_t)S_LEN * DMODEL;     // head-major [16][4096][64]
    float* val_out  = key_out + (size_t)S_LEN * DMODEL;

    unsigned short* Qb = (unsigned short*)d_ws;         // 8 MB each
    unsigned short* Kb = Qb + (size_t)S_LEN * DMODEL;
    unsigned short* Vb = Kb + (size_t)S_LEN * DMODEL;
    unsigned short* Ob = Vb + (size_t)S_LEN * DMODEL;

    dim3 gproj(S_LEN / 128, DMODEL / 128);              // 32 x 8 blocks
    proj_kernel<<<gproj, 256, 0, stream>>>(q, nullptr, Wq, bq, Qb, nullptr, nullptr);
    proj_kernel<<<gproj, 256, 0, stream>>>(k, nullptr, Wk, bk, Kb, key_out, nullptr);
    proj_kernel<<<gproj, 256, 0, stream>>>(v, nullptr, Wv, bv, Vb, val_out, nullptr);

    attn_kernel<<<dim3(S_LEN / 64, NHEADS), 128, 0, stream>>>(Qb, Kb, Vb, Ob);

    proj_kernel<<<gproj, 256, 0, stream>>>(nullptr, Ob, Wo, bo, nullptr, nullptr, out);
}
